// WeatherGate_56298431316292
// MI455X (gfx1250) — matrix operand order repaired and checked by reading, not compiled, not run on hardware
//
#include <hip/hip_runtime.h>
#include <hip/hip_bf16.h>
#include <math.h>

typedef __attribute__((ext_vector_type(16))) _Float16 v16h;
typedef __attribute__((ext_vector_type(8)))  float    v8f;

#define IMG_B   16
#define IMG_H   1024
#define IMG_W   1280
#define NPIX    (IMG_H * IMG_W)      // 1,310,720
#define NGRP    (NPIX / 4)           // 327,680 float4 groups (row-aligned: 1280%4==0)
#define MAXV    16383.0f
#define BINS    256

// ---------------- workspace layout (all offsets in bytes from d_ws) ----------
// [0   .. 512)  : double sums[16][4]   {sum, sumsq, lap_sum, lap_sumsq}
// [512 .. 640)  : uint  minmax[16][2]  {min_bits, max_bits}  (non-negative floats)
// [640 .. 17024): float hist[16][256]
#define WS_SUMS(ws)   ((double*)(ws))
#define WS_MM(ws)     ((unsigned*)((char*)(ws) + 512))
#define WS_HIST(ws)   ((float*)((char*)(ws) + 640))

// -------------------------------------------------------------------- init --
__global__ void wg_init_ws(double* sums, unsigned* minmax, float* hist) {
    int tid = blockIdx.x * blockDim.x + threadIdx.x;
    if (tid < 64) sums[tid] = 0.0;
    if (tid < 32) minmax[tid] = (tid & 1) ? 0u : 0x7F800000u; // min=+inf bits, max=0
    for (int i = tid; i < IMG_B * BINS; i += blockDim.x * gridDim.x) hist[i] = 0.0f;
}

// ---------------------------------------------------- pass A: streaming stats
// float4-vectorized: per 4-pixel group -> 3x b128 (center/up/down) + 2x b32
// edge loads; horizontal neighbors come from registers.
__global__ void __launch_bounds__(256)
wg_stats(const float* __restrict__ image,
         double* __restrict__ sums, unsigned* __restrict__ minmax) {
    const int img = blockIdx.y;
    const float* __restrict__ p = image + (size_t)img * NPIX;
    const float4* __restrict__ p4 = (const float4*)p;
    const int tid    = threadIdx.x;
    const int idx0   = blockIdx.x * blockDim.x + tid;
    const int stride = gridDim.x * blockDim.x;

    double s = 0.0, s2 = 0.0, ls = 0.0, ls2 = 0.0;
    float  mn = 3.4e38f, mx = 0.0f;

    for (int g = idx0; g < NGRP; g += stride) {
        // prefetch next grid-stride group of the center row (global_prefetch_b8)
        if (g + stride < NGRP) __builtin_prefetch(&p4[g + stride], 0, 1);

        const int i = g * 4;                // first pixel of group (row-aligned)
        const int y = i / IMG_W;
        const int x = i - y * IMG_W;        // 0..1276, multiple of 4

        const float4 c = p4[g];
        float4 up, dn;
        if (y > 0)         up = p4[g - IMG_W / 4]; else up = make_float4(0.f, 0.f, 0.f, 0.f);
        if (y < IMG_H - 1) dn = p4[g + IMG_W / 4]; else dn = make_float4(0.f, 0.f, 0.f, 0.f);
        const float lf = (x > 0)             ? p[i - 1] : 0.0f;   // left of pixel 0
        const float rt = (x + 4 < IMG_W)     ? p[i + 4] : 0.0f;   // right of pixel 3

        const float cv[4] = {c.x, c.y, c.z, c.w};
        const float uv[4] = {up.x, up.y, up.z, up.w};
        const float dv[4] = {dn.x, dn.y, dn.z, dn.w};
        const float lv[4] = {lf, c.x, c.y, c.z};
        const float rv[4] = {c.y, c.z, c.w, rt};

#pragma unroll
        for (int j = 0; j < 4; ++j) {
            const float v = cv[j];
            s  += (double)v;
            s2 += (double)v * (double)v;
            mn = fminf(mn, v);
            mx = fmaxf(mx, v);
            const float lap = uv[j] + dv[j] + lv[j] + rv[j] - 4.0f * v;
            ls  += (double)lap;
            ls2 += (double)lap * (double)lap;
        }
    }

    __shared__ double rs[256], rs2[256], rls[256], rls2[256];
    __shared__ float  rmn[256], rmx[256];
    rs[tid] = s; rs2[tid] = s2; rls[tid] = ls; rls2[tid] = ls2;
    rmn[tid] = mn; rmx[tid] = mx;
    __syncthreads();
    for (int off = 128; off > 0; off >>= 1) {
        if (tid < off) {
            rs[tid]  += rs[tid + off];
            rs2[tid] += rs2[tid + off];
            rls[tid] += rls[tid + off];
            rls2[tid]+= rls2[tid + off];
            rmn[tid]  = fminf(rmn[tid], rmn[tid + off]);
            rmx[tid]  = fmaxf(rmx[tid], rmx[tid + off]);
        }
        __syncthreads();
    }
    if (tid == 0) {
        atomicAdd(&sums[img * 4 + 0], rs[0]);
        atomicAdd(&sums[img * 4 + 1], rs2[0]);
        atomicAdd(&sums[img * 4 + 2], rls[0]);
        atomicAdd(&sums[img * 4 + 3], rls2[0]);
        atomicMin(&minmax[img * 2 + 0], __float_as_uint(rmn[0]));
        atomicMax(&minmax[img * 2 + 1], __float_as_uint(rmx[0]));
    }
}

// --------------------------------------------- pass B: triangular soft hist --
// x = clip((v-min)/rng,0,1); p = x*256-0.5; scatter w0->floor(p), w1->floor(p)+1
// float4-vectorized reads (image is L2-resident after pass A); LDS histogram.
__global__ void __launch_bounds__(256)
wg_hist(const float* __restrict__ image,
        const unsigned* __restrict__ minmax, float* __restrict__ hist) {
    const int img = blockIdx.y;
    __shared__ float lh[BINS];
    const int tid = threadIdx.x;
    lh[tid] = 0.0f;
    __syncthreads();

    const float4* __restrict__ p4 = (const float4*)(image + (size_t)img * NPIX);
    const float fmin = __uint_as_float(minmax[img * 2 + 0]);
    const float fmax = __uint_as_float(minmax[img * 2 + 1]);
    float rng = fmax - fmin;
    if (rng == 0.0f) rng = 1.0f;
    const float inv = 1.0f / rng;

    const int idx0   = blockIdx.x * blockDim.x + tid;
    const int stride = gridDim.x * blockDim.x;
    for (int g = idx0; g < NGRP; g += stride) {
        const float4 c = p4[g];
        const float cv[4] = {c.x, c.y, c.z, c.w};
#pragma unroll
        for (int j = 0; j < 4; ++j) {
            float x  = fminf(fmaxf((cv[j] - fmin) * inv, 0.0f), 1.0f);
            float pp = x * (float)BINS - 0.5f;
            float f0 = floorf(pp);
            float w1 = pp - f0;
            int   i0 = (int)f0;            // in [-1, 255]
            if (i0 >= 0)       atomicAdd(&lh[i0], 1.0f - w1);
            if (i0 + 1 <= 255) atomicAdd(&lh[i0 + 1], w1);
        }
    }
    __syncthreads();
    atomicAdd(&hist[img * BINS + tid], lh[tid]);
}

// ------------------------------------------------ head: feats + WMMA MLP ----
// block = 128 threads = 4 waves; wave w owns N-tile [16w, 16w+16) of each layer.
__global__ void __launch_bounds__(128)
wg_head(const double* __restrict__ sums, const unsigned* __restrict__ minmax,
        const float* __restrict__ hist,
        const float* __restrict__ W1, const float* __restrict__ b1,
        const float* __restrict__ W2, const float* __restrict__ b2,
        const float* __restrict__ W3, const float* __restrict__ b3,
        float* __restrict__ out) {
    __shared__ float xA[16][8];     // feats padded to K=8 (cols 5..7 = 0)
    __shared__ float h1[16][64];
    __shared__ float h2[16][64];
    __shared__ float lg[16][3];
    __shared__ float sScale;

    const int tid = threadIdx.x;

    if (tid == 0) {
        float g = 0.0f;
        for (int i = 0; i < IMG_B; ++i) g = fmaxf(g, __uint_as_float(minmax[i * 2 + 1]));
        sScale = (g > 1.0f) ? (1.0f / MAXV) : 1.0f;
    }
    __syncthreads();
    const float sc = sScale;

    // ---- features: one image per thread (tid < 16) ----
    if (tid < 16) {
        const int img = tid;
        const double N = (double)NPIX;
        double s   = sums[img * 4 + 0], s2  = sums[img * 4 + 1];
        double lsm = sums[img * 4 + 2], ls2 = sums[img * 4 + 3];
        float var  = (float)((s2  - s   * s   / N) / (N - 1.0)) * sc * sc;
        float lvar = (float)((ls2 - lsm * lsm / N) / (N - 1.0)) * sc * sc;
        float mn = __uint_as_float(minmax[img * 2 + 0]) * sc;
        float mx = __uint_as_float(minmax[img * 2 + 1]) * sc;

        float hs = 0.0f;
        for (int b = 0; b < BINS; ++b) hs += hist[img * BINS + b];
        const float hinv = 1.0f / (hs + 1e-10f);
        float ent = 0.0f;
        for (int b = 0; b < BINS; ++b) {
            float h = hist[img * BINS + b] * hinv;
            ent -= h * log2f(h + 1e-10f);
        }
        ent *= 0.125f;   // / log2(256)

        xA[img][0] = var; xA[img][1] = mn; xA[img][2] = mx;
        xA[img][3] = ent; xA[img][4] = lvar;
        xA[img][5] = 0.0f; xA[img][6] = 0.0f; xA[img][7] = 0.0f;

        out[64 + img * 5 + 0] = var;
        out[64 + img * 5 + 1] = mn;
        out[64 + img * 5 + 2] = mx;
        out[64 + img * 5 + 3] = ent;
        out[64 + img * 5 + 4] = lvar;
    }
    __syncthreads();

    const int wave = tid >> 5;
    const int lane = tid & 31;
    const int Mrow = lane & 15;
    const int kbA  = (lane < 16) ? 0 : 8;    // unused: A uses kbB (k = 16h + i), as B does
    const int kbB  = (lane < 16) ? 0 : 16;   // B-matrix lane-half K base
    const int ncol = wave * 16 + (lane & 15);
    const int rbase = (lane < 16) ? 0 : 8;   // C/D row base for this lane half

    v16h a, bm;

    // ---- layer 1: h1 = relu(feats @ W1^T + b1), K=32 (zero-padded from 5) ----
    {
        v8f acc = {};
#pragma unroll
        for (int i = 0; i < 16; ++i) {
            int r  = i >> 1;
            int kA = kbB + i;
            a[i]  = (_Float16)((kA < 8) ? xA[Mrow][kA] : 0.0f);
            int kB = kbB + i;
            bm[i] = (_Float16)((kB < 5) ? W1[ncol * 5 + kB] : 0.0f);
        }
        acc = __builtin_amdgcn_wmma_f32_16x16x32_f16(false, a, false, bm,
                                                     (short)0, acc, false, false);
        float bias = b1[ncol];
#pragma unroll
        for (int r = 0; r < 8; ++r)
            h1[rbase + r][ncol] = fmaxf(acc[r] + bias, 0.0f);
    }
    __syncthreads();

    // ---- layer 2: h2 = relu(h1 @ W2^T + b2), K=64 -> two chained WMMAs ----
    {
        v8f acc = {};
#pragma unroll
        for (int koff = 0; koff < 64; koff += 32) {
#pragma unroll
            for (int i = 0; i < 16; ++i) {
                int r  = i >> 1;
                int kA = koff + kbB + i;
                a[i]  = (_Float16)h1[Mrow][kA];
                int kB = koff + kbB + i;
                bm[i] = (_Float16)W2[ncol * 64 + kB];
            }
            acc = __builtin_amdgcn_wmma_f32_16x16x32_f16(false, a, false, bm,
                                                         (short)0, acc, false, false);
        }
        float bias = b2[ncol];
#pragma unroll
        for (int r = 0; r < 8; ++r)
            h2[rbase + r][ncol] = fmaxf(acc[r] + bias, 0.0f);
    }
    __syncthreads();

    // ---- layer 3: logits = h2 @ W3^T + b3 (3 valid columns), wave 0 only ----
    if (wave == 0) {
        const int nco = lane & 15;
        v8f acc = {};
#pragma unroll
        for (int koff = 0; koff < 64; koff += 32) {
#pragma unroll
            for (int i = 0; i < 16; ++i) {
                int r  = i >> 1;
                int kA = koff + kbB + i;
                a[i]  = (_Float16)h2[Mrow][kA];
                int kB = koff + kbB + i;
                bm[i] = (_Float16)((nco < 3) ? W3[nco * 64 + kB] : 0.0f);
            }
            acc = __builtin_amdgcn_wmma_f32_16x16x32_f16(false, a, false, bm,
                                                         (short)0, acc, false, false);
        }
        if (nco < 3) {
            float bias = b3[nco];
#pragma unroll
            for (int r = 0; r < 8; ++r)
                lg[rbase + r][nco] = acc[r] + bias;
        }
    }
    __syncthreads();

    // ---- softmax + argmax, outputs ----
    if (tid < 16) {
        float l0 = lg[tid][0], l1 = lg[tid][1], l2 = lg[tid][2];
        float m  = fmaxf(l0, fmaxf(l1, l2));
        float e0 = expf(l0 - m), e1 = expf(l1 - m), e2 = expf(l2 - m);
        float si = 1.0f / (e0 + e1 + e2);
        out[16 + tid * 3 + 0] = e0 * si;
        out[16 + tid * 3 + 1] = e1 * si;
        out[16 + tid * 3 + 2] = e2 * si;
        int   cls  = 0;
        float best = l0;
        if (l1 > best) { best = l1; cls = 1; }
        if (l2 > best) { best = l2; cls = 2; }
        out[tid] = (float)cls;
    }
}

// -------------------------------------------------------------------- launch
extern "C" void kernel_launch(void* const* d_in, const int* in_sizes, int n_in,
                              void* d_out, int out_size, void* d_ws, size_t ws_size,
                              hipStream_t stream) {
    const float* image = (const float*)d_in[0];
    const float* W1    = (const float*)d_in[1];
    const float* b1    = (const float*)d_in[2];
    const float* W2    = (const float*)d_in[3];
    const float* b2    = (const float*)d_in[4];
    const float* W3    = (const float*)d_in[5];
    const float* b3    = (const float*)d_in[6];
    float* out = (float*)d_out;

    double*   sums = WS_SUMS(d_ws);
    unsigned* mm   = WS_MM(d_ws);
    float*    hist = WS_HIST(d_ws);

    wg_init_ws<<<16, 256, 0, stream>>>(sums, mm, hist);
    wg_stats  <<<dim3(640, IMG_B), 256, 0, stream>>>(image, sums, mm);
    wg_hist   <<<dim3(640, IMG_B), 256, 0, stream>>>(image, mm, hist);
    wg_head   <<<1, 128, 0, stream>>>(sums, mm, hist, W1, b1, W2, b2, W3, b3, out);
}
